// SpGraphAttentionLayer_30099130810500
// MI455X (gfx1250) — matrix operand order repaired and checked by reading, not compiled, not run on hardware
//
#include <hip/hip_runtime.h>
#include <hip/hip_bf16.h>

typedef __attribute__((ext_vector_type(16))) __bf16 v16bf;
typedef __attribute__((ext_vector_type(8)))  float  v8f;

#define ALPHA 0.2f
#define DAMP  0.85f

// exact round-to-nearest-even (used only in one-time weight pre-swizzle)
__device__ __forceinline__ unsigned short f2bf_bits(float f) {
    unsigned u = __float_as_uint(f);
    unsigned r = u + 0x7fffu + ((u >> 16) & 1u);
    return (unsigned short)(r >> 16);
}
// fast path: round-half-up bias + single v_perm_b32 packs two bf16
__device__ __forceinline__ unsigned pack_bf16(float x, float y) {
    unsigned ux = __float_as_uint(x) + 0x8000u;
    unsigned uy = __float_as_uint(y) + 0x8000u;
    // out = { uy[31:16], ux[31:16] } : sel 2,3 -> S1(ux) bytes 2,3 ; 6,7 -> S0(uy)
    return __builtin_amdgcn_perm(uy, ux, 0x07060302u);
}
__device__ __forceinline__ unsigned short f2bf_fast(float f) {
    return (unsigned short)((__float_as_uint(f) + 0x8000u) >> 16);
}
__device__ __forceinline__ float bf2f(unsigned short u) {
    return __uint_as_float(((unsigned)u) << 16);
}
__device__ __forceinline__ float lrelu(float x) { return x > 0.f ? x : ALPHA * x; }
__device__ __forceinline__ float fix0(float x)  { return x == 0.f ? 1e-12f : x; }

// ---------------------------------------------------------------- init
__global__ void init_kernel(float* __restrict__ hprime,   // d_out[0..N*64)
                            float* __restrict__ e_rowsum,
                            float* __restrict__ denom,
                            float* __restrict__ num_rowsum,
                            int*   __restrict__ max_bits,
                            float* __restrict__ sumexp,
                            int N) {
    long i = (long)blockIdx.x * blockDim.x + threadIdx.x;
    long total = (long)N * 64;
    long stride = (long)gridDim.x * blockDim.x;
    for (long j = i; j < total; j += stride) hprime[j] = 0.f;
    if (i < N) { e_rowsum[i] = 0.f; denom[i] = 0.f; num_rowsum[i] = 0.f; }
    if (i == 0) { *max_bits = 0; *sumexp = 0.f; }
}

// ------------------------------------------- pre-swizzle a -> B fragments
// layout: frag[((nt*6+kt)*32 + lane)*16 + 2*v + h]  == B[k][n] in bf16,
// with n = nt*16 + (lane&15), k = kt*32 + (lane>>4)*16 + 2*v + h
__global__ void build_bfrag_kernel(const float* __restrict__ a,  // [64,192]
                                   unsigned short* __restrict__ frag) {
    int idx = blockIdx.x * blockDim.x + threadIdx.x;
    if (idx >= 24 * 32 * 16) return;
    int j    = idx & 15;
    int lane = (idx >> 4) & 31;
    int t    = idx >> 9;            // tile id 0..23
    int nt = t / 6, kt = t % 6;
    int v = j >> 1, h = j & 1;
    int n  = nt * 16 + (lane & 15);
    int k  = kt * 32 + ((lane >> 4) << 4) + (v << 1) + h;
    frag[idx] = f2bf_bits(a[n * 192 + k]);   // B[k][n] = a.T[k][n] = a[n][k]
}

// ---------------------------------------------------------- main WMMA pass
// one wave per 16 edges: C[16 edges x 64 out] = A[16x192] * B[192x64]
// B fragments staged once per block in LDS.
__global__ void edge_gemm_kernel(const float* __restrict__ input,     // [N,64]
                                 const int*   __restrict__ edge,      // [2,E]
                                 const float* __restrict__ eembed,    // [E,64]
                                 const float* __restrict__ a2,        // [64]
                                 const unsigned short* __restrict__ bfrag,
                                 unsigned short* __restrict__ edge_m, // [E,64] bf16
                                 float* __restrict__ edge_e,          // [E]
                                 float* __restrict__ e_rowsum,        // [N]
                                 int E) {
    __shared__ unsigned short sB[24 * 512];   // 24 KB: all B fragments
    {
        const uint4* g4 = (const uint4*)bfrag;
        uint4* s4 = (uint4*)sB;
        for (int i = threadIdx.x; i < 24 * 512 / 8; i += blockDim.x)  // 1536 x 16B
            s4[i] = g4[i];
    }
    __syncthreads();

    int wave = (blockIdx.x * blockDim.x + threadIdx.x) >> 5;
    int lane = threadIdx.x & 31;
    long tileBase = (long)wave * 16;
    if (tileBase >= E) return;                // wave-uniform: EXEC stays all-ones

    int m16  = lane & 15;
    int half = lane >> 4;
    long e = tileBase + m16;                  // row this lane feeds into A
    int s = edge[e];
    int d = edge[(long)E + e];

    const float* seg0 = input  + (long)s * 64;
    const float* seg1 = input  + (long)s * 64 + 32;
    const float* seg2 = input  + (long)d * 64;
    const float* seg3 = input  + (long)d * 64 + 32;
    const float* seg4 = eembed + e * 64;
    const float* seg5 = eembed + e * 64 + 32;
    const float* segs[6] = { seg0, seg1, seg2, seg3, seg4, seg5 };

    v8f acc0 = {}, acc1 = {}, acc2 = {}, acc3 = {};
    #pragma unroll
    for (int kt = 0; kt < 6; ++kt) {
        const float* base = segs[kt];
        // A-fragment 16x32 bf16: lane's 16 K-values are one contiguous
        // 16-float chunk at [half*16, +16).
        float4 c0a = *(const float4*)(base + (half << 4));
        float4 c0b = *(const float4*)(base + (half << 4) + 4);
        float4 c1a = *(const float4*)(base + (half << 4) + 8);
        float4 c1b = *(const float4*)(base + (half << 4) + 12);
        union { v16bf v; unsigned u[8]; } af;
        af.u[0] = pack_bf16(c0a.x, c0a.y);
        af.u[1] = pack_bf16(c0a.z, c0a.w);
        af.u[2] = pack_bf16(c0b.x, c0b.y);
        af.u[3] = pack_bf16(c0b.z, c0b.w);
        af.u[4] = pack_bf16(c1a.x, c1a.y);
        af.u[5] = pack_bf16(c1a.z, c1a.w);
        af.u[6] = pack_bf16(c1b.x, c1b.y);
        af.u[7] = pack_bf16(c1b.z, c1b.w);

        const unsigned short* bp = sB + ((long)kt * 32 + lane) * 16;  // nt=0 slot
        v16bf b0 = *(const v16bf*)(bp);
        v16bf b1 = *(const v16bf*)(bp +  6 * 32 * 16);
        v16bf b2 = *(const v16bf*)(bp + 12 * 32 * 16);
        v16bf b3 = *(const v16bf*)(bp + 18 * 32 * 16);
        acc0 = __builtin_amdgcn_wmma_f32_16x16x32_bf16(false, af.v, false, b0, (short)0, acc0, false, false);
        acc1 = __builtin_amdgcn_wmma_f32_16x16x32_bf16(false, af.v, false, b1, (short)0, acc1, false, false);
        acc2 = __builtin_amdgcn_wmma_f32_16x16x32_bf16(false, af.v, false, b2, (short)0, acc2, false, false);
        acc3 = __builtin_amdgcn_wmma_f32_16x16x32_bf16(false, af.v, false, b3, (short)0, acc3, false, false);
    }

    // C layout: VGPR r, lane -> row (r + 8*half), col (nt*16 + m16)
    float a2v0 = a2[m16], a2v1 = a2[16 + m16], a2v2 = a2[32 + m16], a2v3 = a2[48 + m16];
    float partial[8];
    #pragma unroll
    for (int r = 0; r < 8; ++r) {
        long row = tileBase + (long)half * 8 + r;
        float x0 = lrelu(acc0[r]);
        float x1 = lrelu(acc1[r]);
        float x2 = lrelu(acc2[r]);
        float x3 = lrelu(acc3[r]);
        edge_m[row * 64 +  0 + m16] = f2bf_fast(x0);
        edge_m[row * 64 + 16 + m16] = f2bf_fast(x1);
        edge_m[row * 64 + 32 + m16] = f2bf_fast(x2);
        edge_m[row * 64 + 48 + m16] = f2bf_fast(x3);
        partial[r] = x0 * a2v0 + x1 * a2v1 + x2 * a2v2 + x3 * a2v3;
    }
    // reduce across the 16 lanes of each half -> full dot(edge_m[row], a2)
    #pragma unroll
    for (int mask = 1; mask < 16; mask <<= 1) {
        #pragma unroll
        for (int r = 0; r < 8; ++r)
            partial[r] += __shfl_xor(partial[r], mask, 32);
    }
    if (m16 == 0) {
        #pragma unroll
        for (int r = 0; r < 8; ++r) {
            long row = tileBase + (long)half * 8 + r;
            float ee = __expf(-lrelu(partial[r]));     // exp(powers)
            edge_e[row] = ee;
            atomicAdd(e_rowsum + edge[(long)E + row], ee);
        }
    }
}

// ------------------------------------------------ per-edge scalar passes
__global__ void rel_denom_kernel(const int* __restrict__ edge,
                                 const float* __restrict__ edge_e,
                                 const float* __restrict__ e_rowsum,
                                 float* __restrict__ denom, int E) {
    long i = (long)blockIdx.x * blockDim.x + threadIdx.x;
    if (i >= E) return;
    int s = edge[i];
    atomicAdd(denom + s, edge_e[i] / fix0(e_rowsum[s]));
}

__global__ void num_kernel(const int* __restrict__ edge,
                           const float* __restrict__ edge_e,
                           const float* __restrict__ e_rowsum,
                           const float* __restrict__ denom,
                           const float* __restrict__ entity_rank,
                           float* __restrict__ num_rowsum, int E) {
    long i = (long)blockIdx.x * blockDim.x + threadIdx.x;
    if (i >= E) return;
    int s = edge[i];
    int d = edge[(long)E + i];
    float rel = edge_e[i] / fix0(e_rowsum[s]);
    atomicAdd(num_rowsum + d, rel * entity_rank[s] / fix0(denom[s]));
}

__global__ void rank_kernel(const float* __restrict__ num_rowsum,
                            float* __restrict__ rank_new,
                            float* __restrict__ out_rank,   // d_out[N*64..]
                            int* __restrict__ max_bits, int N) {
    int i = blockIdx.x * blockDim.x + threadIdx.x;
    float r = 0.f;                                   // rank_new is always > 0
    if (i < N) {
        r = (1.0f - DAMP) + DAMP * num_rowsum[i];
        rank_new[i] = r;
        out_rank[i] = r;
    }
    __shared__ float smax[256];
    smax[threadIdx.x] = r;
    __syncthreads();
    for (int s = 128; s > 0; s >>= 1) {
        if (threadIdx.x < s) smax[threadIdx.x] = fmaxf(smax[threadIdx.x], smax[threadIdx.x + s]);
        __syncthreads();
    }
    if (threadIdx.x == 0) atomicMax(max_bits, __float_as_int(smax[0]));  // positive floats
}

__global__ void sumexp_kernel(const float* __restrict__ rank_new,
                              const int* __restrict__ max_bits,
                              float* __restrict__ sumexp, int N) {
    int i = blockIdx.x * blockDim.x + threadIdx.x;
    float mx = __int_as_float(*max_bits);
    float v = (i < N) ? __expf(rank_new[i] - mx) : 0.f;
    __shared__ float ssum[256];
    ssum[threadIdx.x] = v;
    __syncthreads();
    for (int s = 128; s > 0; s >>= 1) {
        if (threadIdx.x < s) ssum[threadIdx.x] += ssum[threadIdx.x + s];
        __syncthreads();
    }
    if (threadIdx.x == 0) atomicAdd(sumexp, ssum[0]);
}

// one thread handles 4 features of one edge (src/dst loads amortized 4x)
__global__ void scatter_kernel(const int* __restrict__ edge,
                               const float* __restrict__ edge_e,
                               const unsigned short* __restrict__ edge_m, // bf16
                               const float* __restrict__ rank_new,
                               const int* __restrict__ max_bits,
                               const float* __restrict__ sumexp,
                               float* __restrict__ hprime, int E) {
    long tid = (long)blockIdx.x * blockDim.x + threadIdx.x;
    if (tid >= (long)E * 16) return;
    long e = tid >> 4;
    int  q = (int)(tid & 15);          // feature quad
    int s = edge[e];
    int d = edge[(long)E + e];
    float mx = __int_as_float(*max_bits);
    float w = (__expf(rank_new[s] - mx) / *sumexp) * edge_e[e];   // softmax[src]*edge_e
    ushort4 m4 = *(const ushort4*)(edge_m + e * 64 + q * 4);
    float* hp = hprime + (long)d * 64 + q * 4;
    atomicAdd(hp + 0, w * bf2f(m4.x));
    atomicAdd(hp + 1, w * bf2f(m4.y));
    atomicAdd(hp + 2, w * bf2f(m4.z));
    atomicAdd(hp + 3, w * bf2f(m4.w));
}

__global__ void finalize_kernel(float* __restrict__ hprime,
                                const float* __restrict__ e_rowsum, int N) {
    long idx = (long)blockIdx.x * blockDim.x + threadIdx.x;
    if (idx >= (long)N * 64) return;
    float h = hprime[idx] / fix0(e_rowsum[idx >> 6]);
    hprime[idx] = h > 0.f ? h : __expf(h) - 1.f;                  // elu
}

// ----------------------------------------------------------------- launch
extern "C" void kernel_launch(void* const* d_in, const int* in_sizes, int n_in,
                              void* d_out, int out_size, void* d_ws, size_t ws_size,
                              hipStream_t stream) {
    const float* input       = (const float*)d_in[0];
    const int*   edge        = (const int*)  d_in[1];
    const float* edge_embed  = (const float*)d_in[2];
    const float* entity_rank = (const float*)d_in[3];
    const float* a           = (const float*)d_in[4];
    const float* a2          = (const float*)d_in[5];

    const int N = in_sizes[0] / 64;
    const int E = in_sizes[1] / 2;

    float* out_h    = (float*)d_out;                // [N,64]
    float* out_rank = (float*)d_out + (long)N * 64; // [N]

    // workspace carve-up (256B aligned pieces)
    char* ws = (char*)d_ws;
    size_t off = 0;
    auto take = [&](size_t bytes) { char* p = ws + off; off = (off + bytes + 255) & ~(size_t)255; return p; };
    unsigned short* bfrag = (unsigned short*)take(24 * 32 * 16 * sizeof(unsigned short));
    float* edge_e     = (float*)take((size_t)E * sizeof(float));
    unsigned short* edge_m = (unsigned short*)take((size_t)E * 64 * sizeof(unsigned short));
    float* e_rowsum   = (float*)take((size_t)N * sizeof(float));
    float* denom      = (float*)take((size_t)N * sizeof(float));
    float* num_rowsum = (float*)take((size_t)N * sizeof(float));
    float* rank_new   = (float*)take((size_t)N * sizeof(float));
    int*   max_bits   = (int*)  take(sizeof(int));
    float* sumexp     = (float*)take(sizeof(float));

    const int BT = 256;
    int nBlocksN = (N + BT - 1) / BT;

    init_kernel<<<nBlocksN, BT, 0, stream>>>(out_h, e_rowsum, denom, num_rowsum, max_bits, sumexp, N);
    build_bfrag_kernel<<<(24 * 32 * 16 + BT - 1) / BT, BT, 0, stream>>>(a, bfrag);

    long waves = (E + 15) / 16;
    long gemmThreads = waves * 32;
    edge_gemm_kernel<<<(int)((gemmThreads + BT - 1) / BT), BT, 0, stream>>>(
        input, edge, edge_embed, a2, bfrag, edge_m, edge_e, e_rowsum, E);

    int nBlocksE = (E + BT - 1) / BT;
    rel_denom_kernel<<<nBlocksE, BT, 0, stream>>>(edge, edge_e, e_rowsum, denom, E);
    num_kernel<<<nBlocksE, BT, 0, stream>>>(edge, edge_e, e_rowsum, denom, entity_rank, num_rowsum, E);
    rank_kernel<<<nBlocksN, BT, 0, stream>>>(num_rowsum, rank_new, out_rank, max_bits, N);
    sumexp_kernel<<<nBlocksN, BT, 0, stream>>>(rank_new, max_bits, sumexp, N);

    long scatterThreads = (long)E * 16;
    scatter_kernel<<<(int)((scatterThreads + BT - 1) / BT), BT, 0, stream>>>(
        edge, edge_e, edge_m, rank_new, max_bits, sumexp, out_h, E);

    long finThreads = (long)N * 64;
    finalize_kernel<<<(int)((finThreads + BT - 1) / BT), BT, 0, stream>>>(out_h, e_rowsum, N);
}
